// DeformConv1D_12833362280764
// MI455X (gfx1250) — matrix operand order repaired and checked by reading, not compiled, not run on hardware
//
#include <hip/hip_runtime.h>
#include <stdint.h>

#define B_   16
#define C_   256
#define OC_  256
#define T_   4096
#define KS_  3
#define TP_  (T_ + 2)
#define TT   128                 // time tile per workgroup in main kernel
#define CCH  32                  // c-chunk (K-chunk) size
#define NCHUNK (KS_ * (C_ / CCH))// 24 K-chunks (n-major)
#define LBS  20                  // LDS B row stride (dwords): 16 k2 + 4 pad, 16B aligned

typedef __attribute__((ext_vector_type(16))) __bf16   v16bf;
typedef __attribute__((ext_vector_type(8)))  float    v8f;
typedef __attribute__((ext_vector_type(4)))  uint32_t u32x4;
typedef __attribute__((ext_vector_type(2), aligned(4))) float f32x2a; // 4B-aligned b64 load

union FragB { uint32_t u[8]; u32x4 q[2]; v16bf v; };

__device__ __forceinline__ uint32_t bf16_rne(float f) {
  uint32_t u = __float_as_uint(f);
  return (u + 0x7FFFu + ((u >> 16) & 1u)) >> 16;
}
__device__ __forceinline__ float bf16_to_f(uint32_t h) {
  return __uint_as_float(h << 16);
}

// ---------------- Kernel A: offset-predicting conv (tiny) ----------------
__global__ __launch_bounds__(128) void offs_kernel(
    const float* __restrict__ x, const float* __restrict__ w_p,
    const float* __restrict__ b_p, float* __restrict__ offs) {
  __shared__ float xs[TT + 2];
  const int tid = threadIdx.x;           // 0..127 == t within tile
  const int t0  = blockIdx.x * TT;
  const int b   = blockIdx.y;
  float a0 = 0.f, a1 = 0.f, a2 = 0.f;
  for (int c = 0; c < C_; ++c) {
    const float* xr = x + ((size_t)b * C_ + c) * T_;
    int q = t0 - 1 + tid;
    xs[tid] = (q >= 0 && q < T_) ? xr[q] : 0.f;
    if (tid < 2) {
      int q2 = t0 - 1 + TT + tid;
      xs[TT + tid] = (q2 >= 0 && q2 < T_) ? xr[q2] : 0.f;
    }
    __syncthreads();
    float xm = xs[tid], x0 = xs[tid + 1], xp = xs[tid + 2];
    const float* wp = w_p + c * KS_;     // w_p[n][c][k] : n*C_*KS_ + c*KS_ + k
    a0 += xm * wp[0]            + x0 * wp[1]            + xp * wp[2];
    a1 += xm * wp[C_*KS_ + 0]   + x0 * wp[C_*KS_ + 1]   + xp * wp[C_*KS_ + 2];
    a2 += xm * wp[2*C_*KS_ + 0] + x0 * wp[2*C_*KS_ + 1] + xp * wp[2*C_*KS_ + 2];
    __syncthreads();
  }
  float* o = offs + ((size_t)b * T_ + (t0 + tid)) * KS_;
  o[0] = a0 + b_p[0];
  o[1] = a1 + b_p[1];
  o[2] = a2 + b_p[2];
}

// ---- Kernel W: pack w_conv into bf16 hi/lo, A-fragment order [n][cc][o][h][v] ----
__global__ __launch_bounds__(256) void pack_w(
    const float* __restrict__ w, uint32_t* __restrict__ wphi,
    uint32_t* __restrict__ wplo) {
  int idx = blockIdx.x * 256 + threadIdx.x;
  if (idx >= KS_ * 8 * OC_ * 2 * 8) return;     // 3*8*256*2*8 = 98304 dwords
  int v  = idx & 7;
  int h  = (idx >> 3) & 1;
  int o  = (idx >> 4) & (OC_ - 1);
  int cc = (idx >> 12) & 7;
  int n  = idx >> 15;
  // A k order used here (contiguous, same as B): lane l, VGPR v -> k2 = v + ((l>>4)<<3)
  int k2rel = v + (h << 3);
  int c2 = cc * 16 + k2rel;                      // global c-pair index
  // w_conv[o][c][n] : o*C_*KS_ + c*KS_ + n
  float we = w[(size_t)o * C_ * KS_ + (2 * c2)     * KS_ + n];
  float wo = w[(size_t)o * C_ * KS_ + (2 * c2 + 1) * KS_ + n];
  uint32_t he = bf16_rne(we), ho = bf16_rne(wo);
  float re = we - bf16_to_f(he), ro = wo - bf16_to_f(ho);
  wphi[idx] = he | (ho << 16);
  wplo[idx] = bf16_rne(re) | (bf16_rne(ro) << 16);
}

// ---------------- Kernel B: fused gather-interp + bf16x3 WMMA GEMM ----------------
__global__ __launch_bounds__(256) void deform_gemm(
    const float* __restrict__ x, const float* __restrict__ offs,
    const uint32_t* __restrict__ wphi, const uint32_t* __restrict__ wplo,
    float* __restrict__ out) {
  __shared__ uint32_t lbh[TT * LBS];   // B-chunk hi: [t][k2], k2-pairs packed
  __shared__ uint32_t lbl[TT * LBS];   // B-chunk lo
  __shared__ float    loff[TT * KS_];

  const int tid  = threadIdx.x;
  const int lane = tid & 31;
  const int wv   = tid >> 5;           // 0..7, owns M rows [32*wv, 32*wv+32)
  const int b    = blockIdx.y;
  const int t0   = blockIdx.x * TT;

  for (int i = tid; i < TT * KS_; i += 256)
    loff[i] = offs[((size_t)b * T_ + t0) * KS_ + i];
  __syncthreads();

  // builder thread: fixed t column (tb), half selects K sub-range
  const int tb   = tid & (TT - 1);
  const int half = tid >> 7;
  int ilA[KS_], irA[KS_];
  float glA[KS_], grA[KS_];
#pragma unroll
  for (int n = 0; n < KS_; ++n) {
    // p = (t+1) + (n-1) + offset
    float p  = (float)(t0 + tb + n) + loff[tb * KS_ + n];
    float ql = floorf(p);
    float qr = ql + 1.f;
    ql = fminf(fmaxf(ql, 0.f), (float)(TP_ - 1));
    qr = fminf(fmaxf(qr, 0.f), (float)(TP_ - 1));
    p  = fminf(fmaxf(p,  0.f), (float)(TP_ - 1));
    glA[n] = p - ql;  grA[n] = qr - p;
    ilA[n] = (int)ql; irA[n] = (int)qr;
  }

  const u32x4* wph4 = (const u32x4*)wphi;
  const u32x4* wpl4 = (const u32x4*)wplo;

  v8f acc[16];
#pragma unroll
  for (int i = 0; i < 16; ++i) acc[i] = (v8f){0.f,0.f,0.f,0.f,0.f,0.f,0.f,0.f};

  for (int chunk = 0; chunk < NCHUNK; ++chunk) {
    const int n  = chunk >> 3;   // K is n-major: same interp weights per chunk
    const int cc = chunk & 7;

    // ---- cooperative build: gather + lerp + bf16 split -> LDS (b128 stores) ----
    {
      const int   il = ilA[n], ir = irA[n];
      const float gl = glA[n], gr = grA[n];
      const float* xb = x + (size_t)b * C_ * T_;
      const int cbase = cc * CCH + half * 16;
      float vE[8], vO[8];
      if ((il >= 1) & (ir <= T_) & (ir == il + 1)) {
        // interior (~99.95% of threads): both taps adjacent -> single b64 per row
#pragma unroll
        for (int j = 0; j < 8; ++j) {
          const float* xe = xb + (size_t)(cbase + 2 * j) * T_ + (il - 1);
          f32x2a e2 = *(const f32x2a*)xe;
          f32x2a o2 = *(const f32x2a*)(xe + T_);
          vE[j] = gr * e2[0] + gl * e2[1];
          vO[j] = gr * o2[0] + gl * o2[1];
        }
      } else {
        // boundary: exact reference clamping/zero-pad semantics
        const bool okl = (il >= 1) && (il <= T_);
        const bool okr = (ir >= 1) && (ir <= T_);
#pragma unroll
        for (int j = 0; j < 8; ++j) {
          const float* xe = xb + (size_t)(cbase + 2 * j) * T_;
          const float* xo = xe + T_;
          float vel = okl ? xe[il - 1] : 0.f;
          float ver = okr ? xe[ir - 1] : 0.f;
          float vol = okl ? xo[il - 1] : 0.f;
          float vor = okr ? xo[ir - 1] : 0.f;
          vE[j] = gr * vel + gl * ver;
          vO[j] = gr * vol + gl * vor;
        }
      }
      u32x4 h0, h1, l0, l1;
#pragma unroll
      for (int j = 0; j < 8; ++j) {
        uint32_t he = bf16_rne(vE[j]), ho = bf16_rne(vO[j]);
        float re = vE[j] - bf16_to_f(he), ro = vO[j] - bf16_to_f(ho);
        uint32_t dh = he | (ho << 16);
        uint32_t dl = bf16_rne(re) | (bf16_rne(ro) << 16);
        if (j < 4) { h0[j] = dh; l0[j] = dl; }
        else       { h1[j - 4] = dh; l1[j - 4] = dl; }
      }
      const int wbase = tb * LBS + half * 8;
      *(u32x4*)&lbh[wbase]     = h0;
      *(u32x4*)&lbh[wbase + 4] = h1;
      *(u32x4*)&lbl[wbase]     = l0;
      *(u32x4*)&lbl[wbase + 4] = l1;
    }
    __syncthreads();

    // ---- A fragments: two fully-coalesced b128 loads per frag ----
    FragB ah[2], al[2];
    {
      const int obase = wv * 32 + (lane & 15);
      const int hsel  = lane >> 4;
#pragma unroll
      for (int mt = 0; mt < 2; ++mt) {
        size_t qi = ((size_t)((n * 8 + cc) * 256 + obase + mt * 16) * 2 + hsel) * 2;
        ah[mt].q[0] = wph4[qi];
        ah[mt].q[1] = wph4[qi + 1];
        al[mt].q[0] = wpl4[qi];
        al[mt].q[1] = wpl4[qi + 1];
      }
    }

    // ---- WMMA: acc += Ah*Bh + Ah*Bl + Al*Bh (bf16x3 ~ fp32) ----
    const int kb = (lane >> 4) << 3;     // k2 base: lanes16-31 take K=16..31
#pragma unroll
    for (int nt = 0; nt < 8; ++nt) {
      const int rb = (nt * 16 + (lane & 15)) * LBS + kb;
      FragB bh, bl;
      bh.q[0] = *(const u32x4*)&lbh[rb];
      bh.q[1] = *(const u32x4*)&lbh[rb + 4];
      bl.q[0] = *(const u32x4*)&lbl[rb];
      bl.q[1] = *(const u32x4*)&lbl[rb + 4];
#pragma unroll
      for (int mt = 0; mt < 2; ++mt) {
        v8f a = acc[mt * 8 + nt];
        a = __builtin_amdgcn_wmma_f32_16x16x32_bf16(false, ah[mt].v, false, bh.v, (short)0, a, false, false);
        a = __builtin_amdgcn_wmma_f32_16x16x32_bf16(false, ah[mt].v, false, bl.v, (short)0, a, false, false);
        a = __builtin_amdgcn_wmma_f32_16x16x32_bf16(false, al[mt].v, false, bh.v, (short)0, a, false, false);
        acc[mt * 8 + nt] = a;
      }
    }
    __syncthreads();
  }

  // ---- epilogue: C/D layout -> out[b][o][t], 64B-coalesced per half-wave ----
  const int tcol0 = t0 + (lane & 15);
  const int orow0 = wv * 32 + ((lane >> 4) << 3);
#pragma unroll
  for (int mt = 0; mt < 2; ++mt)
#pragma unroll
    for (int nt = 0; nt < 8; ++nt) {
      v8f a = acc[mt * 8 + nt];
      int t = tcol0 + nt * 16;
#pragma unroll
      for (int i = 0; i < 8; ++i) {
        int o = orow0 + mt * 16 + i;
        out[((size_t)b * OC_ + o) * T_ + t] = a[i];
      }
    }
}

extern "C" void kernel_launch(void* const* d_in, const int* in_sizes, int n_in,
                              void* d_out, int out_size, void* d_ws, size_t ws_size,
                              hipStream_t stream) {
  const float* x      = (const float*)d_in[0];
  const float* w_p    = (const float*)d_in[1];
  const float* b_p    = (const float*)d_in[2];
  const float* w_conv = (const float*)d_in[3];
  float* out = (float*)d_out;

  // workspace: offsets [B][T][3] f32, then packed-w hi/lo [3][8][256][2][8] dwords
  float*    offs = (float*)d_ws;
  size_t    offs_n = (size_t)B_ * T_ * KS_;
  uint32_t* wphi = (uint32_t*)(offs + offs_n);
  uint32_t* wplo = wphi + (size_t)KS_ * 8 * OC_ * 2 * 8;

  dim3 gA(T_ / TT, B_);
  offs_kernel<<<gA, 128, 0, stream>>>(x, w_p, b_p, offs);

  int nW = KS_ * 8 * OC_ * 2 * 8;
  pack_w<<<(nW + 255) / 256, 256, 0, stream>>>(w_conv, wphi, wplo);

  dim3 gB(T_ / TT, B_);
  deform_gemm<<<gB, 256, 0, stream>>>(x, offs, wphi, wplo, out);
}
